// MHSA_81466939670561
// MI455X (gfx1250) — compile-verified, matrix operand order checked by reading, not run on hardware
//
#include <hip/hip_runtime.h>
#include <hip/hip_bf16.h>

typedef __bf16 bf16;
typedef __attribute__((ext_vector_type(16))) bf16  v16bf;
typedef __attribute__((ext_vector_type(8)))  bf16  v8bf;
typedef __attribute__((ext_vector_type(8)))  float v8f;

#define DIM   256
#define HEADS 4
#define HDIM  64
#define BATCH 8
#define SEQ   2048
#define MTOK  (BATCH * SEQ)      /* 16384 tokens */
#define SCALE 0.125f             /* 64^-0.5 */
#define KBLK  32                 /* keys per attention iteration */

/* padded LDS leading dims (elements) to spread fragment reads across banks */
#define KLD 72
#define VLD 40
#define PLD 40

/* ---- fragment loaders -------------------------------------------------- */
/* 16-bit A-matrix 16x32 layout (ISA 7.12.2):
   lane L: row = L&15, kgroup = L>>4.
   a[0..7]  = 8 contiguous elems at col  kgroup*8
   a[8..15] = 8 contiguous elems at col  16 + kgroup*8
   B-matrix 32x16 for C = A * B with B supplied as Bt[n][k] row-major uses the
   identical pattern with "row" meaning output column n.                      */
__device__ __forceinline__ v16bf load_frag(const bf16* p, int ld) {
    int lane = threadIdx.x & 31;
    int row  = lane & 15;
    int kg   = lane >> 4;
    const bf16* base = p + row * ld + kg * 8;
    v8bf lo = *(const v8bf*)(base);
    v8bf hi = *(const v8bf*)(base + 16);
    v16bf a;
#pragma unroll
    for (int i = 0; i < 8; ++i) { a[i] = lo[i]; a[i + 8] = hi[i]; }
    return a;
}

__device__ __forceinline__ v8f wmma_bf16(v16bf a, v16bf b, v8f c) {
    return __builtin_amdgcn_wmma_f32_16x16x32_bf16(
        /*neg_a=*/false, a, /*neg_b=*/false, b,
        /*c_mod=*/(short)0, c, /*reuse_a=*/false, /*reuse_b=*/false);
}

/* reductions within the 16-lane groups that share a C-fragment row */
__device__ __forceinline__ float redmax16(float v) {
#pragma unroll
    for (int m = 8; m >= 1; m >>= 1) v = fmaxf(v, __shfl_xor(v, m, 32));
    return v;
}
__device__ __forceinline__ float redsum16(float v) {
#pragma unroll
    for (int m = 8; m >= 1; m >>= 1) v += __shfl_xor(v, m, 32);
    return v;
}

/* async global -> LDS 16B copy (per-lane), tracked with ASYNCcnt */
__device__ __forceinline__ void async_cp16(uint32_t lds_off, const bf16* gaddr) {
    asm volatile("global_load_async_to_lds_b128 %0, %1, off"
                 :: "v"(lds_off), "v"(gaddr) : "memory");
}

/* ---- fp32 -> bf16 conversion ------------------------------------------- */
__global__ void cvt_f32_bf16(const float* __restrict__ in, bf16* __restrict__ out, int n) {
    int i = blockIdx.x * blockDim.x + threadIdx.x;
    if (i < n) out[i] = (bf16)in[i];
}

/* ---- fused QKV projection: [16384,256] x [768,256]^T + bias ------------ */
__global__ void __launch_bounds__(256) qkv_gemm(
        const bf16* __restrict__ X, const bf16* __restrict__ W,
        const float* __restrict__ bias,
        bf16* __restrict__ Q, bf16* __restrict__ K, bf16* __restrict__ Vt) {
    int w    = threadIdx.x >> 5;
    int lane = threadIdx.x & 31;
    int m0   = blockIdx.x * 32 + (w & 1) * 16;   /* token tile  */
    int o0   = blockIdx.y * 64 + (w >> 1) * 16;  /* out-feature tile */

    v8f acc = {};
#pragma unroll
    for (int kk = 0; kk < DIM; kk += 32) {
        v16bf a = load_frag(X + (size_t)m0 * DIM + kk, DIM);
        v16bf b = load_frag(W + (size_t)o0 * DIM + kk, DIM);
        acc = wmma_bf16(a, b, acc);
    }

    int hi   = lane >> 4;
    int col  = o0 + (lane & 15);
    float bv = bias[col];
    int three = col >> 8;          /* 0:Q 1:K 2:V  (o = t*256 + h*64 + d) */
    int head  = (col >> 6) & 3;
    int dcol  = col & 63;

#pragma unroll
    for (int r = 0; r < 8; ++r) {
        int m    = m0 + r + 8 * hi;
        int bidx = m >> 11;
        int n    = m & (SEQ - 1);
        bf16 hv  = (bf16)(acc[r] + bv);
        if (three == 0)
            Q[(((size_t)bidx * HEADS + head) * SEQ + n) * HDIM + dcol] = hv;
        else if (three == 1)
            K[(((size_t)bidx * HEADS + head) * SEQ + n) * HDIM + dcol] = hv;
        else /* V stored transposed: [b,h,d,n] */
            Vt[(((size_t)bidx * HEADS + head) * HDIM + dcol) * SEQ + n] = hv;
    }
}

/* ---- flash-style attention --------------------------------------------
   One block (8 waves) owns 128 query rows of one (batch,head).
   K (32x64) and V^T (64x32) tiles are double-buffered in LDS via async
   global->LDS copies shared by all 8 waves; each wave computes a 16-row
   S tile (4 WMMAs), does online softmax in registers, transposes P through
   a per-wave LDS tile, and accumulates O with 4 more WMMAs.               */
__global__ void __launch_bounds__(256) attn_kernel(
        const bf16* __restrict__ Q, const bf16* __restrict__ Kt,
        const bf16* __restrict__ Vt, bf16* __restrict__ AO) {
    __shared__ alignas(16) bf16 sK[2][KBLK * KLD];   /* 2 x 4.5 KB */
    __shared__ alignas(16) bf16 sV[2][HDIM * VLD];   /* 2 x 5.0 KB */
    __shared__ alignas(16) bf16 sP[8][16 * PLD];     /* 8 x 1.25 KB */

    int t    = threadIdx.x;
    int w    = t >> 5;
    int lane = t & 31;
    int hi   = lane >> 4;

    int bh   = blockIdx.x >> 4;                  /* batch*head, 0..31 */
    int tile = blockIdx.x & 15;
    int qrow = tile * 128 + w * 16;

    const bf16* Qh = Q  + (size_t)bh * SEQ * HDIM;
    const bf16* Kh = Kt + (size_t)bh * SEQ * HDIM;
    const bf16* Vh = Vt + (size_t)bh * HDIM * SEQ;

    /* per-thread async staging: 256 threads x 16B = one 4KB tile each     */
    int krow = t >> 3, kchk = t & 7;             /* K: 32 rows x 128B      */
    int vrow = t >> 2, vchk = t & 3;             /* V: 64 rows x 64B       */
    uint32_t ldsK0 = (uint32_t)(uintptr_t)(&sK[0][0]) + (uint32_t)(krow * KLD + kchk * 8) * 2u;
    uint32_t ldsK1 = (uint32_t)(uintptr_t)(&sK[1][0]) + (uint32_t)(krow * KLD + kchk * 8) * 2u;
    uint32_t ldsV0 = (uint32_t)(uintptr_t)(&sV[0][0]) + (uint32_t)(vrow * VLD + vchk * 8) * 2u;
    uint32_t ldsV1 = (uint32_t)(uintptr_t)(&sV[1][0]) + (uint32_t)(vrow * VLD + vchk * 8) * 2u;

    v16bf q0 = load_frag(Qh + (size_t)qrow * HDIM,      HDIM);
    v16bf q1 = load_frag(Qh + (size_t)qrow * HDIM + 32, HDIM);

    v8f o0 = {}, o1 = {}, o2 = {}, o3 = {};
    float mrow[8], lrow[8];
#pragma unroll
    for (int r = 0; r < 8; ++r) { mrow[r] = -3.0e38f; lrow[r] = 0.0f; }

    const float SC2 = SCALE * 1.44269504088896340736f;  /* scale * log2(e) */
    bf16* pw = &sP[w][0];

    /* prologue: stage first K/V tile into buffer 0 */
    async_cp16(ldsK0, Kh + (size_t)krow * HDIM + kchk * 8);
    async_cp16(ldsV0, Vh + (size_t)vrow * SEQ  + vchk * 8);

    for (int j = 0; j < SEQ; j += KBLK) {
        int cur = (j >> 5) & 1;
        if (j + KBLK < SEQ) {   /* prefetch next tile into the other buffer */
            int jn = j + KBLK;
            async_cp16(cur ? ldsK0 : ldsK1, Kh + (size_t)(jn + krow) * HDIM + kchk * 8);
            async_cp16(cur ? ldsV0 : ldsV1, Vh + (size_t)vrow * SEQ + jn + vchk * 8);
            asm volatile("s_wait_asynccnt 0x2" ::: "memory");
        } else {
            asm volatile("s_wait_asynccnt 0x0" ::: "memory");
        }
        __syncthreads();                  /* all waves' tile data visible */

        const bf16* Ks = &sK[cur][0];
        const bf16* Vs = &sV[cur][0];

        /* S block 16x32 : 4 WMMAs, operands from LDS */
        v16bf kA0 = load_frag(Ks,                 KLD);
        v16bf kA1 = load_frag(Ks + 32,            KLD);
        v16bf kB0 = load_frag(Ks + 16 * KLD,      KLD);
        v16bf kB1 = load_frag(Ks + 16 * KLD + 32, KLD);
        v8f s0 = {}, s1 = {};
        s0 = wmma_bf16(q0, kA0, s0);
        s0 = wmma_bf16(q1, kA1, s0);
        s1 = wmma_bf16(q0, kB0, s1);
        s1 = wmma_bf16(q1, kB1, s1);

        /* online softmax (base-2 domain), rescale O, stage P into LDS */
#pragma unroll
        for (int r = 0; r < 8; ++r) {
            float a = s0[r] * SC2;
            float b = s1[r] * SC2;
            float mx   = redmax16(fmaxf(a, b));
            float mnew = fmaxf(mrow[r], mx);
            float alpha = __builtin_amdgcn_exp2f(mrow[r] - mnew);
            float p0 = __builtin_amdgcn_exp2f(a - mnew);
            float p1 = __builtin_amdgcn_exp2f(b - mnew);
            float rs = redsum16(p0 + p1);
            lrow[r] = lrow[r] * alpha + rs;
            mrow[r] = mnew;
            o0[r] *= alpha; o1[r] *= alpha; o2[r] *= alpha; o3[r] *= alpha;
            int row = r + 8 * hi;
            pw[row * PLD + (lane & 15)]      = (bf16)p0;
            pw[row * PLD + 16 + (lane & 15)] = (bf16)p1;
        }
        /* same-wave DS ops are in-order: dscnt fence makes all lanes'
           stores visible before the transposed re-load */
        asm volatile("s_wait_dscnt 0" ::: "memory");

        v16bf pa  = load_frag(pw, PLD);
        v16bf vb0 = load_frag(Vs,            VLD);
        v16bf vb1 = load_frag(Vs + 16 * VLD, VLD);
        v16bf vb2 = load_frag(Vs + 32 * VLD, VLD);
        v16bf vb3 = load_frag(Vs + 48 * VLD, VLD);
        o0 = wmma_bf16(pa, vb0, o0);
        o1 = wmma_bf16(pa, vb1, o1);
        o2 = wmma_bf16(pa, vb2, o2);
        o3 = wmma_bf16(pa, vb3, o3);

        __syncthreads();   /* everyone done reading cur before it refills */
    }

    /* epilogue: normalize, write head-major into [b, n, h*64+d] (bf16) */
    int bidx = bh >> 2;
    int head = bh & 3;
    int c    = lane & 15;
#pragma unroll
    for (int r = 0; r < 8; ++r) {
        int n = qrow + r + 8 * hi;
        float inv = 1.0f / lrow[r];
        size_t base = ((size_t)bidx * SEQ + n) * DIM + head * HDIM;
        AO[base +      c] = (bf16)(o0[r] * inv);
        AO[base + 16 + c] = (bf16)(o1[r] * inv);
        AO[base + 32 + c] = (bf16)(o2[r] * inv);
        AO[base + 48 + c] = (bf16)(o3[r] * inv);
    }
}

/* ---- output projection: [16384,256] x [256,256]^T + bias -> fp32 ------- */
__global__ void __launch_bounds__(256) proj_gemm(
        const bf16* __restrict__ X, const bf16* __restrict__ W,
        const float* __restrict__ bias, float* __restrict__ out) {
    int w    = threadIdx.x >> 5;
    int lane = threadIdx.x & 31;
    int m0   = blockIdx.x * 32 + (w & 1) * 16;
    int o0   = blockIdx.y * 64 + (w >> 1) * 16;

    v8f acc = {};
#pragma unroll
    for (int kk = 0; kk < DIM; kk += 32) {
        v16bf a = load_frag(X + (size_t)m0 * DIM + kk, DIM);
        v16bf b = load_frag(W + (size_t)o0 * DIM + kk, DIM);
        acc = wmma_bf16(a, b, acc);
    }

    int hi  = lane >> 4;
    int col = o0 + (lane & 15);
    float bv = bias[col];
#pragma unroll
    for (int r = 0; r < 8; ++r) {
        size_t m = (size_t)(m0 + r + 8 * hi);
        out[m * DIM + col] = acc[r] + bv;
    }
}

/* ---- host side --------------------------------------------------------- */
extern "C" void kernel_launch(void* const* d_in, const int* in_sizes, int n_in,
                              void* d_out, int out_size, void* d_ws, size_t ws_size,
                              hipStream_t stream) {
    const float* x       = (const float*)d_in[0];
    const float* Wqkv_f  = (const float*)d_in[1];
    const float* bqkv    = (const float*)d_in[2];
    const float* Wproj_f = (const float*)d_in[3];
    const float* bproj   = (const float*)d_in[4];
    float* out = (float*)d_out;

    char* ws = (char*)d_ws;
    bf16* Xbf = (bf16*)(ws);                       /* 16384*256*2 = 8388608  */
    bf16* Wq  = (bf16*)(ws + 8388608);             /*   768*256*2 =  393216  */
    bf16* Wp  = (bf16*)(ws + 8781824);             /*   256*256*2 =  131072  */
    bf16* Qb  = (bf16*)(ws + 8912896);             /* 8*4*2048*64*2 = 8388608*/
    bf16* Kb  = (bf16*)(ws + 17301504);
    bf16* Vtb = (bf16*)(ws + 25690112);
    bf16* AOb = (bf16*)(ws + 34078720);            /* ends at 42467328 bytes */

    const int nX  = MTOK * DIM;         /* 4194304 */
    const int nWq = 3 * DIM * DIM;      /*  196608 */
    const int nWp = DIM * DIM;          /*   65536 */
    cvt_f32_bf16<<<(nX  + 255) / 256, 256, 0, stream>>>(x,       Xbf, nX);
    cvt_f32_bf16<<<(nWq + 255) / 256, 256, 0, stream>>>(Wqkv_f,  Wq,  nWq);
    cvt_f32_bf16<<<(nWp + 255) / 256, 256, 0, stream>>>(Wproj_f, Wp,  nWp);

    qkv_gemm  <<<dim3(MTOK / 32, (3 * DIM) / 64), 256, 0, stream>>>(Xbf, Wq, bqkv, Qb, Kb, Vtb);
    attn_kernel<<<BATCH * HEADS * (SEQ / 128),     256, 0, stream>>>(Qb, Kb, Vtb, AOb);
    proj_gemm <<<dim3(MTOK / 32, DIM / 64),        256, 0, stream>>>(AOb, Wp, bproj, out);
}
